// MQOTLayer_38345468019362
// MI455X (gfx1250) — matrix operand order repaired and checked by reading, not compiled, not run on hardware
//
#include <hip/hip_runtime.h>
#include <math.h>

#define B_    8
#define TA    1024
#define TV    512
#define DD    768
#define HH    4
#define NIT   20
#define TOL_  0.001f

typedef __attribute__((ext_vector_type(16))) _Float16 v16h;
typedef __attribute__((ext_vector_type(8)))  _Float16 v8h;
typedef __attribute__((ext_vector_type(8)))  float    v8f;

__device__ __forceinline__ float wave_max(float x) {
    #pragma unroll
    for (int o = 16; o > 0; o >>= 1) x = fmaxf(x, __shfl_xor(x, o, 32));
    return x;
}
__device__ __forceinline__ float wave_sum(float x) {
    #pragma unroll
    for (int o = 16; o > 0; o >>= 1) x += __shfl_xor(x, o, 32);
    return x;
}
__device__ __forceinline__ float get_eps(const float* p_le) {
    return fmaxf(log1pf(expf(*p_le)), 0.005f);   // max(softplus, EPS_MIN)
}

// ---------------- init: zero u, v, diff slots, done flag -------------------
__global__ void k_init(unsigned* p, int n) {
    int i = blockIdx.x * blockDim.x + threadIdx.x;
    if (i < n) p[i] = 0u;
}

// ---------------- L2 row normalize f32 -> f16 (row length 768) -------------
__global__ void k_norm(const float* __restrict__ src, _Float16* __restrict__ dst, int nrows) {
    int wid  = (blockIdx.x * blockDim.x + threadIdx.x) >> 5;
    int lane = threadIdx.x & 31;
    if (wid >= nrows) return;
    const float* r = src + (size_t)wid * DD;
    float vals[24];
    float ss = 0.f;
    #pragma unroll
    for (int t = 0; t < 24; ++t) { float x = r[lane + 32 * t]; vals[t] = x; ss += x * x; }
    ss = wave_sum(ss);
    float inv = 1.0f / fmaxf(sqrtf(ss), 1e-12f);
    _Float16* o = dst + (size_t)wid * DD;
    #pragma unroll
    for (int t = 0; t < 24; ++t) o[lane + 32 * t] = (_Float16)(vals[t] * inv);
}

// ---------------- time-bias MLP: (i/Ta, j/Tv) -> gelu -> H outputs ---------
__global__ void k_tb(const float* __restrict__ W1, const float* __restrict__ b1,
                     const float* __restrict__ W2, const float* __restrict__ b2,
                     float* __restrict__ tb) {
    int idx = blockIdx.x * blockDim.x + threadIdx.x;
    if (idx >= TA * TV) return;
    int i = idx / TV, j = idx % TV;
    float x0 = (float)i / (float)TA;
    float x1 = (float)j / (float)TV;
    float acc[HH];
    #pragma unroll
    for (int h = 0; h < HH; ++h) acc[h] = b2[h];
    #pragma unroll 4
    for (int k = 0; k < 64; ++k) {
        float z = fmaf(x0, W1[k], fmaf(x1, W1[64 + k], b1[k]));
        float g = 0.5f * z * (1.0f + erff(z * 0.70710678118654752f)); // exact gelu
        #pragma unroll
        for (int h = 0; h < HH; ++h) acc[h] = fmaf(g, W2[k * HH + h], acc[h]);
    }
    #pragma unroll
    for (int h = 0; h < HH; ++h) tb[(size_t)h * TA * TV + idx] = acc[h];
}

// ---------------- WMMA GEMM: C_feat = 1 - A_n * V_n^T ----------------------
// one wave -> 16(M) x 64(N) tile; A fragment reused over 4 B subtiles
__global__ void k_gemm(const _Float16* __restrict__ a16, const _Float16* __restrict__ v16,
                       float* __restrict__ cf) {
    int wid  = (blockIdx.x * blockDim.x + threadIdx.x) >> 5;  // 4096 waves
    int lane = threadIdx.x & 31;
    int b    = wid >> 9;            // 512 waves / batch
    int rem  = wid & 511;
    int mt   = rem >> 3;            // 0..63
    int nt   = rem & 7;             // 0..7
    int M0 = mt * 16, N0 = nt * 64;
    int lm = lane & 15, hi = lane >> 4;

    const _Float16* Abase = a16 + ((size_t)(b * TA + M0 + lm)) * DD + hi * 8;
    const _Float16* Bbase = v16 + ((size_t)(b * TV + N0 + lm)) * DD + hi * 8;

    v8f acc[4] = {};
    for (int kb = 0; kb < DD; kb += 32) {
        v16h a;
        {
            v8h lo = *(const v8h*)(Abase + kb);        // K = kb + 8*hi + 0..7
            v8h hh = *(const v8h*)(Abase + kb + 16);   // K = kb + 16 + 8*hi + 0..7
            #pragma unroll
            for (int t = 0; t < 8; ++t) { a[t] = lo[t]; a[t + 8] = hh[t]; }
        }
        #pragma unroll
        for (int s = 0; s < 4; ++s) {
            const _Float16* bp = Bbase + (size_t)s * 16 * DD + kb; // K = kb + 8*hi + 0..7, kb + 16 + 8*hi + 0..7
            v16h bb;
            {
                v8h lo = *(const v8h*)bp;
                v8h hh = *(const v8h*)(bp + 16);
                #pragma unroll
                for (int t = 0; t < 8; ++t) { bb[t] = lo[t]; bb[t + 8] = hh[t]; }
            }
            acc[s] = __builtin_amdgcn_wmma_f32_16x16x32_f16(
                false, a, false, bb, (short)0, acc[s], false, false);
        }
    }
    #pragma unroll
    for (int s = 0; s < 4; ++s) {
        int N = N0 + s * 16 + lm;
        #pragma unroll
        for (int r = 0; r < 8; ++r) {
            int M = M0 + r + 8 * hi;
            cf[((size_t)b * TA + M) * TV + N] = 1.0f - acc[s][r];
        }
    }
}

// ---------------- Sinkhorn u-update: row LSE over j (one wave / row) -------
__global__ void k_u(const float* __restrict__ cf, const float* __restrict__ tb,
                    const float* __restrict__ q,
                    const float* __restrict__ p_le, const float* __restrict__ p_lt,
                    const float* __restrict__ p_lq,
                    const float* __restrict__ vv, float* __restrict__ uu,
                    unsigned* __restrict__ diffslot, const int* __restrict__ done) {
    if (*(volatile const int*)done) return;
    int wid  = (blockIdx.x * blockDim.x + threadIdx.x) >> 5;  // B*H*TA rows
    int lane = threadIdx.x & 31;
    int b = wid / (HH * TA);
    int h = (wid / TA) % HH;
    int i = wid % TA;
    float eps = get_eps(p_le);
    float lt = *p_lt, lq = *p_lq;

    const float* cfr = cf + ((size_t)b * TA + i) * TV;
    const float* tbr = tb + ((size_t)h * TA + i) * TV;
    const float* qr  = q  + (size_t)b * TV;
    const float* vr  = vv + (size_t)(b * HH + h) * TV;

    float x[16];
    float m = -3.402823466e38f;
    #pragma unroll
    for (int t = 0; t < 16; ++t) {
        int j = lane + 32 * t;
        float c  = cfr[j] + lt * tbr[j] + lq * (1.0f - qr[j]);
        float xx = (vr[j] - c) / eps;
        x[t] = xx; m = fmaxf(m, xx);
    }
    m = wave_max(m);
    float s = 0.f;
    #pragma unroll
    for (int t = 0; t < 16; ++t) s += expf(x[t] - m);
    s = wave_sum(s);
    float lse = m + logf(s);
    size_t ui = (size_t)(b * HH + h) * TA + i;
    float uo = uu[ui];
    float un = -lse / 1.1f + uo * (0.1f / 1.1f);
    if (lane == 0) {
        uu[ui] = un;
        atomicMax(diffslot, __float_as_uint(fabsf(un - uo)));
    }
}

// ---------------- Sinkhorn v-update: column online-LSE over i --------------
__global__ void k_v(const float* __restrict__ cf, const float* __restrict__ tb,
                    const float* __restrict__ q,
                    const float* __restrict__ p_le, const float* __restrict__ p_lt,
                    const float* __restrict__ p_lq,
                    const float* __restrict__ uu, float* __restrict__ vv,
                    const int* __restrict__ done) {
    if (*(volatile const int*)done) return;
    int idx = blockIdx.x * blockDim.x + threadIdx.x;  // B*H*TV
    int b = idx / (HH * TV);
    int h = (idx / TV) % HH;
    int j = idx % TV;
    float eps = get_eps(p_le);
    float lt = *p_lt, lq = *p_lq;
    float qp = lq * (1.0f - q[(size_t)b * TV + j]);

    const float* cfc = cf + (size_t)b * TA * TV + j;
    const float* tbc = tb + (size_t)h * TA * TV + j;
    const float* ur  = uu + (size_t)(b * HH + h) * TA;

    float m = -3.402823466e38f, s = 0.f;
    for (int i = 0; i < TA; ++i) {
        float c  = cfc[(size_t)i * TV] + lt * tbc[(size_t)i * TV] + qp;
        float xx = (ur[i] - c) / eps;
        float nm = fmaxf(m, xx);
        s = s * expf(m - nm) + expf(xx - nm);
        m = nm;
    }
    float lse = m + logf(s);
    float vo = vv[idx];
    vv[idx] = -lse / 1.1f + vo * (0.1f / 1.1f);
}

// ---------------- done |= (max|du| < TOL) ----------------------------------
__global__ void k_done(const unsigned* __restrict__ diffslot, int* __restrict__ done) {
    if (__uint_as_float(*diffslot) < TOL_) *done = 1;
}

// ---------------- final: P = exp((-cost+u+v)/eps), row-normalized ----------
__global__ void k_p(const float* __restrict__ cf, const float* __restrict__ tb,
                    const float* __restrict__ q,
                    const float* __restrict__ p_le, const float* __restrict__ p_lt,
                    const float* __restrict__ p_lq,
                    const float* __restrict__ uu, const float* __restrict__ vv,
                    float* __restrict__ out) {
    int wid  = (blockIdx.x * blockDim.x + threadIdx.x) >> 5;  // B*H*TA rows
    int lane = threadIdx.x & 31;
    int b = wid / (HH * TA);
    int h = (wid / TA) % HH;
    int i = wid % TA;
    float eps = get_eps(p_le);
    float lt = *p_lt, lq = *p_lq;

    const float* cfr = cf + ((size_t)b * TA + i) * TV;
    const float* tbr = tb + ((size_t)h * TA + i) * TV;
    const float* qr  = q  + (size_t)b * TV;
    const float* vr  = vv + (size_t)(b * HH + h) * TV;
    float ui = uu[(size_t)(b * HH + h) * TA + i];

    float e[16];
    float s = 0.f;
    #pragma unroll
    for (int t = 0; t < 16; ++t) {
        int j = lane + 32 * t;
        float c  = cfr[j] + lt * tbr[j] + lq * (1.0f - qr[j]);
        float ee = expf((ui + vr[j] - c) / eps);
        e[t] = ee; s += ee;
    }
    s = wave_sum(s);
    float inv = 1.0f / (s + 1e-8f);
    float* orow = out + ((size_t)((b * HH + h) * TA) + i) * TV;
    #pragma unroll
    for (int t = 0; t < 16; ++t) orow[lane + 32 * t] = e[t] * inv;
}

// ---------------------------------------------------------------------------
extern "C" void kernel_launch(void* const* d_in, const int* in_sizes, int n_in,
                              void* d_out, int out_size, void* d_ws, size_t ws_size,
                              hipStream_t stream) {
    const float* audio = (const float*)d_in[0];
    const float* video = (const float*)d_in[1];
    const float* qual  = (const float*)d_in[2];
    const float* p_le  = (const float*)d_in[3];
    const float* p_lt  = (const float*)d_in[4];
    const float* p_lq  = (const float*)d_in[5];
    const float* W1    = (const float*)d_in[6];
    const float* b1    = (const float*)d_in[7];
    const float* W2    = (const float*)d_in[8];
    const float* b2    = (const float*)d_in[9];
    float* out = (float*)d_out;

    char* ws = (char*)d_ws;
    const size_t OFF_A16  = 0;                                  // 8*1024*768*2
    const size_t OFF_V16  = OFF_A16 + (size_t)B_ * TA * DD * 2; // 8*512*768*2
    const size_t OFF_CF   = OFF_V16 + (size_t)B_ * TV * DD * 2; // 8*1024*512*4
    const size_t OFF_TB   = OFF_CF  + (size_t)B_ * TA * TV * 4; // 4*1024*512*4
    const size_t OFF_U    = OFF_TB  + (size_t)HH * TA * TV * 4; // zeroed region starts here
    const size_t OFF_V    = OFF_U   + (size_t)B_ * HH * TA * 4;
    const size_t OFF_DIFF = OFF_V   + (size_t)B_ * HH * TV * 4;
    const size_t OFF_DONE = OFF_DIFF + NIT * 4;

    _Float16* a16 = (_Float16*)(ws + OFF_A16);
    _Float16* v16 = (_Float16*)(ws + OFF_V16);
    float* cf     = (float*)(ws + OFF_CF);
    float* tb     = (float*)(ws + OFF_TB);
    float* uu     = (float*)(ws + OFF_U);
    float* vv     = (float*)(ws + OFF_V);
    unsigned* dif = (unsigned*)(ws + OFF_DIFF);
    int* done     = (int*)(ws + OFF_DONE);

    // zero u, v, diff slots, done flag (contiguous)
    int nzero = B_ * HH * TA + B_ * HH * TV + NIT + 1;
    k_init<<<(nzero + 255) / 256, 256, 0, stream>>>((unsigned*)(ws + OFF_U), nzero);

    // normalize: one wave per 768-length row
    k_norm<<<(B_ * TA) / 8, 256, 0, stream>>>(audio, a16, B_ * TA);
    k_norm<<<(B_ * TV) / 8, 256, 0, stream>>>(video, v16, B_ * TV);

    // time-bias MLP
    k_tb<<<(TA * TV) / 256, 256, 0, stream>>>(W1, b1, W2, b2, tb);

    // WMMA GEMM -> C_feat  (4096 waves, 8 waves/block)
    k_gemm<<<512, 256, 0, stream>>>(a16, v16, cf);

    // Sinkhorn iterations with reference "done"-freeze semantics
    for (int k = 0; k < NIT; ++k) {
        k_u<<<(B_ * HH * TA) / 8, 256, 0, stream>>>(cf, tb, qual, p_le, p_lt, p_lq,
                                                    vv, uu, dif + k, done);
        k_v<<<(B_ * HH * TV) / 256, 256, 0, stream>>>(cf, tb, qual, p_le, p_lt, p_lq,
                                                      uu, vv, done);
        k_done<<<1, 1, 0, stream>>>(dif + k, done);
    }

    // final transport plan, row-normalized
    k_p<<<(B_ * HH * TA) / 8, 256, 0, stream>>>(cf, tb, qual, p_le, p_lt, p_lq,
                                                uu, vv, out);
}
